// Model_49297634623642
// MI455X (gfx1250) — matrix operand order repaired and checked by reading, not compiled, not run on hardware
//
#include <hip/hip_runtime.h>
#include <hip/hip_bf16.h>
#include <math.h>

// ---------------- problem constants (from reference) ----------------
#define N_NODES 10000
#define NFEAT   512
#define NHID    64
#define HEADS   8
#define NEDGE   320000
#define EP      (NEDGE + N_NODES)   // edges + self loops = 330000
#define NEG_SLOPE 0.2f
#define BSTRIDE 72                  // padded LDS row stride (bf16 elems): 144B, 16B-aligned

typedef __attribute__((ext_vector_type(16))) __bf16 v16bf;
typedef __attribute__((ext_vector_type(8)))  __bf16 v8bf;
typedef __attribute__((ext_vector_type(8)))  float  v8f;

static __device__ __forceinline__ __bf16 tobf(float f) { return (__bf16)f; }

// ---- CDNA5 async global->LDS copy (ASYNCcnt-tracked), per ISA 08_async_tensor ----
static __device__ __forceinline__ void async_g2l_b128(const __bf16* g, __bf16* l) {
  unsigned long long gaddr = (unsigned long long)g;
  unsigned lds_off = (unsigned)(unsigned long long)(void*)l;  // flat addr[31:0] == LDS offset
  asm volatile("global_load_async_to_lds_b128 %0, %1, off"
               :: "v"(lds_off), "v"(gaddr) : "memory");
}
static __device__ __forceinline__ void wait_async0() {
  asm volatile("s_wait_asynccnt 0x0" ::: "memory");
}

// A fragment per ISA 16-bit A 16x32 layout: two contiguous 16B chunks per lane
static __device__ __forceinline__ v16bf loadA(const __bf16* Arow, int k) {
  v8bf lo = *(const v8bf*)(Arow + k);
  v8bf hi = *(const v8bf*)(Arow + k + 16);
  return __builtin_shufflevector(lo, hi, 0,1,2,3,4,5,6,7,8,9,10,11,12,13,14,15);
}

// float atomic max via monotonic int mapping (init to -inf)
static __device__ __forceinline__ void atomicMaxF(float* addr, float v) {
  int iv = __float_as_int(v);
  if (iv >= 0) atomicMax((int*)addr, iv);
  else         atomicMin((unsigned int*)addr, (unsigned int)iv);
}

// ---------------- elementwise helpers ----------------
__global__ void k_f2bf(const float* __restrict__ in, __bf16* __restrict__ out, int n) {
  int i = blockIdx.x * blockDim.x + threadIdx.x;
  if (i < n) out[i] = tobf(in[i]);
}

__global__ void k_fill(float* __restrict__ p, float v, int n) {
  int i = blockIdx.x * blockDim.x + threadIdx.x;
  if (i < n) p[i] = v;
}

// ---------------- LDS-staged, double-buffered bf16 WMMA GEMM ----------------
// C[M,Nc] = A[M,K] * B[K,Nc].  Block = 256 threads = 8 waves, block tile 128(M) x 64(N).
// B tile (32K x 64N) async-staged to LDS, shared by all 8 waves, double-buffered.
// A fragment register-double-buffered straight from global.
__global__ void __launch_bounds__(256)
k_gemm_wmma(const __bf16* __restrict__ A, const __bf16* __restrict__ B,
            float* __restrict__ C, int M, int K, int Nc, int nGroups) {
  __shared__ __bf16 Bs[2][32 * BSTRIDE];
  const int tid  = threadIdx.x;
  const int wave = tid >> 5, lane = tid & 31;
  const int mb = blockIdx.x / nGroups;
  const int ng = blockIdx.x - mb * nGroups;
  const int n0 = ng << 6;
  const int mTiles = M >> 4;
  int mt = mb * 8 + wave;
  if (mt >= mTiles) mt = mTiles - 1;      // duplicate work in last block; identical stores

  const int row   = lane & 15;
  const int kbase = (lane >> 4) << 3;     // 0 or 8
  const __bf16* Arow = A + (size_t)(mt * 16 + row) * K;

  // thread tid stages one 16B chunk of the 32x64 B tile: row = tid>>3, col8 = tid&7
  const int br = tid >> 3, bc8 = (tid & 7) << 3;
  const __bf16* Bg0 = B + (size_t)br * Nc + n0 + bc8;
  __bf16* Bl0a = &Bs[0][br * BSTRIDE + bc8];
  __bf16* Bl0b = &Bs[1][br * BSTRIDE + bc8];

  async_g2l_b128(Bg0, Bl0a);              // prologue: stage k0 = 0 into buf 0
  wait_async0();
  __syncthreads();

  v8f acc0 = {}, acc1 = {}, acc2 = {}, acc3 = {};
  v16bf a_cur = loadA(Arow, kbase);
  for (int k0 = 0; k0 < K; k0 += 32) {
    const int buf = (k0 >> 5) & 1;
    const bool more = (k0 + 32) < K;
    if (more)                              // stream next B tile into the other buffer
      async_g2l_b128(Bg0 + (size_t)(k0 + 32) * Nc, buf ? Bl0a : Bl0b);
    v16bf a_next;
    if (more) a_next = loadA(Arow, k0 + 32 + kbase);

    const __bf16* Bl = &Bs[buf][lane & 15];
    v16bf b0, b1, b2, b3;
#pragma unroll
    for (int s = 0; s < 16; ++s) {
      const __bf16* Bp = Bl + (kbase + (s & 7) + ((s >> 3) << 4)) * BSTRIDE;
      b0[s] = Bp[0]; b1[s] = Bp[16]; b2[s] = Bp[32]; b3[s] = Bp[48];
    }
    acc0 = __builtin_amdgcn_wmma_f32_16x16x32_bf16(false, a_cur, false, b0, (short)0, acc0, false, false);
    acc1 = __builtin_amdgcn_wmma_f32_16x16x32_bf16(false, a_cur, false, b1, (short)0, acc1, false, false);
    acc2 = __builtin_amdgcn_wmma_f32_16x16x32_bf16(false, a_cur, false, b2, (short)0, acc2, false, false);
    acc3 = __builtin_amdgcn_wmma_f32_16x16x32_bf16(false, a_cur, false, b3, (short)0, acc3, false, false);
    if (more) { a_cur = a_next; wait_async0(); __syncthreads(); }
  }
  // C/D layout: VGPR v -> M = v + 8*(lane>=16); N = n0 + 16j + (lane&15)
  const int mbase = mt * 16 + ((lane >> 4) << 3);
  const int ncol  = n0 + (lane & 15);
#pragma unroll
  for (int v = 0; v < 8; ++v) {
    size_t r = (size_t)(mbase + v) * Nc;
    C[r + ncol +  0] = acc0[v];
    C[r + ncol + 16] = acc1[v];
    C[r + ncol + 32] = acc2[v];
    C[r + ncol + 48] = acc3[v];
  }
}

// ---------------- GAT attention pieces ----------------
__global__ void k_alpha(const float* __restrict__ h, const float* __restrict__ a_src,
                        const float* __restrict__ a_dst, float* __restrict__ asrc,
                        float* __restrict__ adst, int heads, int C, int n) {
  int i = blockIdx.x * blockDim.x + threadIdx.x;
  if (i >= n * heads) return;
  int node = i / heads, hd = i - node * heads;
  const float* hp = h + (size_t)node * heads * C + hd * C;
  const float* as = a_src + hd * C;
  const float* ad = a_dst + hd * C;
  float s = 0.f, d = 0.f;
  for (int c = 0; c < C; ++c) { float v = hp[c]; s += v * as[c]; d += v * ad[c]; }
  asrc[i] = s; adst[i] = d;
}

static __device__ __forceinline__ void edge_sd(const int* adj, int e, int E, int& s, int& d) {
  if (e < E) { s = adj[e]; d = adj[E + e]; } else { s = d = e - E; }
}

__global__ void k_edge_max(const int* __restrict__ adj, const float* __restrict__ asrc,
                           const float* __restrict__ adst, float* __restrict__ e_out,
                           float* __restrict__ m, int heads, int E, int Ep) {
  int i = blockIdx.x * blockDim.x + threadIdx.x;
  if (i >= Ep * heads) return;
  int e = i / heads, hd = i - e * heads, s, d;
  edge_sd(adj, e, E, s, d);
  float v = asrc[s * heads + hd] + adst[d * heads + hd];
  v = v > 0.f ? v : v * NEG_SLOPE;
  e_out[i] = v;
  atomicMaxF(&m[d * heads + hd], v);
}

__global__ void k_edge_exp(const int* __restrict__ adj, const float* __restrict__ e_in,
                           const float* __restrict__ m, float* __restrict__ ex,
                           float* __restrict__ den, int heads, int E, int Ep) {
  int i = blockIdx.x * blockDim.x + threadIdx.x;
  if (i >= Ep * heads) return;
  int e = i / heads, hd = i - e * heads, s, d;
  edge_sd(adj, e, E, s, d);
  float v = __expf(e_in[i] - m[d * heads + hd]);
  ex[i] = v;
  atomicAdd(&den[d * heads + hd], v);
}

__global__ void k_agg(const int* __restrict__ adj, const float* __restrict__ h,
                      const float* __restrict__ ex, const float* __restrict__ den,
                      float* __restrict__ out, int heads, int C, int E, int Ep) {
  int i = blockIdx.x * blockDim.x + threadIdx.x;
  if (i >= Ep * heads) return;
  int e = i / heads, hd = i - e * heads, s, d;
  edge_sd(adj, e, E, s, d);
  float alpha = ex[i] / den[d * heads + hd];
  const float* hp = h + (size_t)s * heads * C + hd * C;
  float* op = out + (size_t)d * heads * C + hd * C;
  for (int c = 0; c < C; ++c) atomicAdd(&op[c], hp[c] * alpha);
}

__global__ void k_bias_elu_bf(const float* __restrict__ in, const float* __restrict__ b,
                              __bf16* __restrict__ out, int F, int n) {
  int i = blockIdx.x * blockDim.x + threadIdx.x;
  if (i >= n) return;
  float v = in[i] + b[i % F];
  v = v > 0.f ? v : (__expf(v) - 1.f);
  out[i] = tobf(v);
}

__global__ void k_logsoftmax(const float* __restrict__ in, const float* __restrict__ b,
                             float* __restrict__ z, int C, int n) {
  int node = blockIdx.x * blockDim.x + threadIdx.x;
  if (node >= n) return;
  const float* p = in + (size_t)node * C;
  float mx = -INFINITY;
  for (int c = 0; c < C; ++c) { float t = p[c] + b[c]; mx = t > mx ? t : mx; }
  float s = 0.f;
  for (int c = 0; c < C; ++c) s += __expf(p[c] + b[c] - mx);
  float ls = logf(s);
  float* zp = z + (size_t)node * C;
  for (int c = 0; c < C; ++c) zp[c] = p[c] + b[c] - mx - ls;
}

// ---------------- fused decoder: relu([z[row]||z[col]] @ lin1 + b) @ lin2 + b2 ----------------
// lin1 rows 0..63 (64x64 bf16) async-staged once into LDS; one wave per 16-edge tile; 16 WMMAs;
// ReLU + lin2 dot via half-wave shfl reduction.
__global__ void __launch_bounds__(128)
k_decoder(const float* __restrict__ z, const int* __restrict__ adj,
          const __bf16* __restrict__ lin1b, const float* __restrict__ lin1_bias,
          const float* __restrict__ lin2w, const float* __restrict__ lin2b,
          float* __restrict__ out, int E) {
  __shared__ __bf16 L1s[128 * BSTRIDE];
  const int tid = threadIdx.x;
  // stage 64x64 lin1 rows 0..63 (512 x 16B chunks) via async copies
#pragma unroll
  for (int c = tid; c < 512; c += 128) {
    int r = c >> 3, c8 = (c & 7) << 3;
    async_g2l_b128(lin1b + r * NHID + c8, &L1s[r * BSTRIDE + c8]);
  }
  wait_async0();
  __syncthreads();

  const int w = blockIdx.x * (blockDim.x >> 5) + (tid >> 5);
  const int lane  = tid & 31;
  const int row   = lane & 15;
  const int kbase = (lane >> 4) << 3;
  const int et    = w * 16 + row;
  const int r = adj[et], c = adj[E + et];
  const float* zrow = z + (size_t)r * NHID;
  const float* zcol = z + (size_t)c * NHID;

  v8f acc0 = {}, acc1 = {}, acc2 = {}, acc3 = {};
#pragma unroll
  for (int k0 = 0; k0 < 128; k0 += 32) {
    v16bf a;
#pragma unroll
    for (int v = 0; v < 8; ++v) {
      int k = k0 + kbase + ((v & 3) << 1) + ((v >> 2) << 4);   // even, never straddles 64
      float f0 = (k < NHID) ? zrow[k]     : zcol[k - NHID];
      float f1 = (k < NHID) ? zrow[k + 1] : zcol[k + 1 - NHID];
      a[2 * v] = tobf(f0); a[2 * v + 1] = tobf(f1);
    }
    const __bf16* Brow = &L1s[k0 * BSTRIDE + (lane & 15)];
    v16bf b0, b1, b2, b3;
#pragma unroll
    for (int s = 0; s < 16; ++s) {
      const __bf16* Bp = Brow + (kbase + (s & 7) + ((s >> 3) << 4)) * BSTRIDE;
      b0[s] = Bp[0]; b1[s] = Bp[16]; b2[s] = Bp[32]; b3[s] = Bp[48];
    }
    acc0 = __builtin_amdgcn_wmma_f32_16x16x32_bf16(false, a, false, b0, (short)0, acc0, false, false);
    acc1 = __builtin_amdgcn_wmma_f32_16x16x32_bf16(false, a, false, b1, (short)0, acc1, false, false);
    acc2 = __builtin_amdgcn_wmma_f32_16x16x32_bf16(false, a, false, b2, (short)0, acc2, false, false);
    acc3 = __builtin_amdgcn_wmma_f32_16x16x32_bf16(false, a, false, b3, (short)0, acc3, false, false);
  }
  const int ln = lane & 15;
  float s[8];
#pragma unroll
  for (int v = 0; v < 8; ++v) s[v] = 0.f;
  const v8f* accs[4] = { &acc0, &acc1, &acc2, &acc3 };
#pragma unroll
  for (int j = 0; j < 4; ++j) {
    int n  = 16 * j + ln;
    float w2 = lin2w[n], bb = lin1_bias[n];
#pragma unroll
    for (int v = 0; v < 8; ++v) {
      float t = (*accs[j])[v] + bb;
      t = t > 0.f ? t : 0.f;
      s[v] += t * w2;
    }
  }
#pragma unroll
  for (int off = 1; off < 16; off <<= 1)
#pragma unroll
    for (int v = 0; v < 8; ++v) s[v] += __shfl_xor(s[v], off, 32);
  if (ln == 0) {
    int mbase = w * 16 + ((lane >> 4) << 3);
#pragma unroll
    for (int v = 0; v < 8; ++v) out[mbase + v] = s[v] + lin2b[0];
  }
}

// ---------------- launcher ----------------
static inline char* carve(char*& p, size_t bytes) {
  char* r = p;
  p += (bytes + 255) & ~(size_t)255;
  return r;
}

extern "C" void kernel_launch(void* const* d_in, const int* in_sizes, int n_in,
                              void* d_out, int out_size, void* d_ws, size_t ws_size,
                              hipStream_t stream) {
  (void)in_sizes; (void)n_in; (void)out_size; (void)ws_size;
  const float* x       = (const float*)d_in[0];
  const int*   adj     = (const int*)d_in[1];
  const int*   adjd    = (const int*)d_in[2];
  const float* W1      = (const float*)d_in[3];
  const float* a_src1  = (const float*)d_in[4];
  const float* a_dst1  = (const float*)d_in[5];
  const float* b1      = (const float*)d_in[6];
  const float* W2      = (const float*)d_in[7];
  const float* a_src2  = (const float*)d_in[8];
  const float* a_dst2  = (const float*)d_in[9];
  const float* b2      = (const float*)d_in[10];
  const float* lin1_W  = (const float*)d_in[11];
  const float* lin1_b  = (const float*)d_in[12];
  const float* lin2_W  = (const float*)d_in[13];
  const float* lin2_b  = (const float*)d_in[14];
  float* out = (float*)d_out;

  // workspace carve (~95 MB total)
  char* p = (char*)d_ws;
  __bf16* xb    = (__bf16*)carve(p, (size_t)N_NODES * NFEAT * 2);
  __bf16* w1b   = (__bf16*)carve(p, (size_t)NFEAT * HEADS * NHID * 2);
  float*  h1    = (float*) carve(p, (size_t)N_NODES * HEADS * NHID * 4);
  float*  as1   = (float*) carve(p, (size_t)N_NODES * HEADS * 4);
  float*  ad1   = (float*) carve(p, (size_t)N_NODES * HEADS * 4);
  float*  m1    = (float*) carve(p, (size_t)N_NODES * HEADS * 4);
  float*  den1  = (float*) carve(p, (size_t)N_NODES * HEADS * 4);
  float*  e1    = (float*) carve(p, (size_t)EP * HEADS * 4);
  float*  ex1   = (float*) carve(p, (size_t)EP * HEADS * 4);
  float*  out1  = (float*) carve(p, (size_t)N_NODES * HEADS * NHID * 4);
  __bf16* helub = (__bf16*)carve(p, (size_t)N_NODES * HEADS * NHID * 2);
  __bf16* w2b   = (__bf16*)carve(p, (size_t)HEADS * NHID * NHID * 2);
  float*  h2    = (float*) carve(p, (size_t)N_NODES * NHID * 4);
  float*  as2   = (float*) carve(p, (size_t)N_NODES * 4);
  float*  ad2   = (float*) carve(p, (size_t)N_NODES * 4);
  float*  m2    = (float*) carve(p, (size_t)N_NODES * 4);
  float*  den2  = (float*) carve(p, (size_t)N_NODES * 4);
  float*  e2    = (float*) carve(p, (size_t)EP * 4);
  float*  ex2   = (float*) carve(p, (size_t)EP * 4);
  float*  out2  = (float*) carve(p, (size_t)N_NODES * NHID * 4);
  float*  zbuf  = (float*) carve(p, (size_t)N_NODES * NHID * 4);
  __bf16* lin1b = (__bf16*)carve(p, (size_t)2 * NHID * NHID * 2);

  const int T = 256;
  auto blks = [](long n, int t) { return (int)((n + t - 1) / t); };
  const float NEGINF = -INFINITY;

  // convert fp32 -> bf16 GEMM operands
  k_f2bf<<<blks((long)N_NODES * NFEAT, T), T, 0, stream>>>(x, xb, N_NODES * NFEAT);
  k_f2bf<<<blks((long)NFEAT * HEADS * NHID, T), T, 0, stream>>>(W1, w1b, NFEAT * HEADS * NHID);
  k_f2bf<<<blks((long)HEADS * NHID * NHID, T), T, 0, stream>>>(W2, w2b, HEADS * NHID * NHID);
  k_f2bf<<<blks((long)2 * NHID * NHID, T), T, 0, stream>>>(lin1_W, lin1b, 2 * NHID * NHID);

  // init accumulators
  k_fill<<<blks(N_NODES * HEADS, T), T, 0, stream>>>(m1, NEGINF, N_NODES * HEADS);
  k_fill<<<blks(N_NODES * HEADS, T), T, 0, stream>>>(den1, 0.f, N_NODES * HEADS);
  k_fill<<<blks((long)N_NODES * HEADS * NHID, T), T, 0, stream>>>(out1, 0.f, N_NODES * HEADS * NHID);
  k_fill<<<blks(N_NODES, T), T, 0, stream>>>(m2, NEGINF, N_NODES);
  k_fill<<<blks(N_NODES, T), T, 0, stream>>>(den2, 0.f, N_NODES);
  k_fill<<<blks((long)N_NODES * NHID, T), T, 0, stream>>>(out2, 0.f, N_NODES * NHID);

  // ---- conv1: X[10000,512] @ W1[512,512] ----
  {
    int mBlocks = (N_NODES / 16 + 7) / 8;                    // 79
    int nGroups = HEADS * NHID / 64;                         // 8
    k_gemm_wmma<<<mBlocks * nGroups, 256, 0, stream>>>(xb, w1b, h1, N_NODES, NFEAT,
                                                       HEADS * NHID, nGroups);
  }
  k_alpha<<<blks(N_NODES * HEADS, T), T, 0, stream>>>(h1, a_src1, a_dst1, as1, ad1, HEADS, NHID, N_NODES);
  k_edge_max<<<blks((long)EP * HEADS, T), T, 0, stream>>>(adjd, as1, ad1, e1, m1, HEADS, NEDGE, EP);
  k_edge_exp<<<blks((long)EP * HEADS, T), T, 0, stream>>>(adjd, e1, m1, ex1, den1, HEADS, NEDGE, EP);
  k_agg<<<blks((long)EP * HEADS, T), T, 0, stream>>>(adjd, h1, ex1, den1, out1, HEADS, NHID, NEDGE, EP);
  k_bias_elu_bf<<<blks((long)N_NODES * HEADS * NHID, T), T, 0, stream>>>(out1, b1, helub,
                                                                         HEADS * NHID,
                                                                         N_NODES * HEADS * NHID);
  // ---- conv2 (heads=1): H[10000,512] @ W2[512,64] ----
  {
    int mBlocks = (N_NODES / 16 + 7) / 8;                    // 79
    k_gemm_wmma<<<mBlocks, 256, 0, stream>>>(helub, w2b, h2, N_NODES, HEADS * NHID, NHID, 1);
  }
  k_alpha<<<blks(N_NODES, T), T, 0, stream>>>(h2, a_src2, a_dst2, as2, ad2, 1, NHID, N_NODES);
  k_edge_max<<<blks(EP, T), T, 0, stream>>>(adjd, as2, ad2, e2, m2, 1, NEDGE, EP);
  k_edge_exp<<<blks(EP, T), T, 0, stream>>>(adjd, e2, m2, ex2, den2, 1, NEDGE, EP);
  k_agg<<<blks(EP, T), T, 0, stream>>>(adjd, h2, ex2, den2, out2, 1, NHID, NEDGE, EP);
  k_logsoftmax<<<blks(N_NODES, T), T, 0, stream>>>(out2, b2, zbuf, NHID, N_NODES);

  // ---- decoder over `adj` (no self loops) ----
  {
    int waves = NEDGE / 16;                                   // 20000
    k_decoder<<<waves / 4, 128, 0, stream>>>(zbuf, adj, lin1b, lin1_b, lin2_W, lin2_b, out, NEDGE);
  }
}
